// TESNN_9423158247574
// MI455X (gfx1250) — matrix operand order repaired and checked by reading, not compiled, not run on hardware
//
#include <hip/hip_runtime.h>

#define T_STEPS 784
#define HDIM 64
#define OUTD 10
#define NWAVES 4
#define WGSIZE (NWAVES * 32)
#define SPW 16                  // samples per wave (WMMA M)
#define SPWG (NWAVES * SPW)     // samples per workgroup

typedef _Float16 half_t;
typedef _Float16 v16h __attribute__((ext_vector_type(16)));
typedef _Float16 v8h  __attribute__((ext_vector_type(8)));
typedef float    v8f  __attribute__((ext_vector_type(8)));

__device__ __forceinline__ v16h combine16(v8h a, v8h b) {
  v16h r;
#pragma unroll
  for (int i = 0; i < 8; ++i) { r[i] = a[i]; r[i + 8] = b[i]; }
  return r;
}

__device__ __forceinline__ v8f wmma16(v16h a, v16h b, v8f c) {
  // (neg_a, A, neg_b, B, c_mod, C, reuse_a, reuse_b)
  return __builtin_amdgcn_wmma_f32_16x16x32_f16(false, a, false, b, (short)0, c,
                                                false, false);
}

// A fragment (16x32 f16, M=lane%16). Staging is row-major [M][64] f16.
// lanes 0-15: halves0-7 -> K 0-7,  halves8-15 -> K 16-23
// lanes16-31: halves0-7 -> K 8-15, halves8-15 -> K 24-31   (per ISA 7.12.2)
__device__ __forceinline__ v16h load_afrag(const half_t* stg, int lo, int hi, int kt) {
  const half_t* p = stg + lo * HDIM + kt * 32 + hi * 8;
  v8h a = *(const v8h*)p;
  v8h b = *(const v8h*)(p + 16);
  return combine16(a, b);
}

// B fragment (32x16 f16) from row-major f16 weight [j][64]; B[k][n] = W[n][k].
// lane N = n; same K order as the A fragment (halves0-7 -> K 8h..8h+7, halves8-15 -> K 16+8h..).
// The index is made opaque with an empty volatile asm so LICM cannot hoist
// these loop-invariant LDS loads (144 VGPRs of fragments would force spills);
// address-space inference still lowers them as ds_load_b128.
__device__ __forceinline__ v16h load_bfrag(const half_t* w, int lo, int hi, int nt, int kt) {
  int base = (nt * 16 + lo) * HDIM + kt * 32 + hi * 8;
  asm volatile("" : "+v"(base));
  const half_t* p = w + base;
  v8h a = *(const v8h*)p;
  v8h b = *(const v8h*)(p + 16);
  return combine16(a, b);
}

// SNN state update for one 16x16 C-tile. mem in VGPRs, th in per-lane LDS,
// new spike written (as f16) into the wave's staging tile (M x K row-major).
//   th' = th + mm*co - (th-0.5)*0.02 == fma(th, 0.98, fma(mm, co, 0.01))
//   mm' = mm*0.5*(1-sp) + h, sp = (mm>th)  ==  sp ? h : fma(mm, 0.5, h)
__device__ __forceinline__ void update_tile(float* m, float* thl, const float* hin,
                                            float co, half_t* stg, int lo, int hi,
                                            int nt) {
#pragma unroll
  for (int r = 0; r < 8; ++r) {
    float th = thl[nt * 8 + r];
    float mm = m[nt * 8 + r];
    float thn = __builtin_fmaf(th, 0.98f, __builtin_fmaf(mm, co, 0.01f));
    float mhalf = __builtin_fmaf(mm, 0.5f, hin[r]);
    float mn = (mm > th) ? hin[r] : mhalf;      // prev spike == (mem > th)
    float sn = (mn > thn) ? 1.f : 0.f;
    thl[nt * 8 + r] = thn;
    m[nt * 8 + r] = mn;
    stg[(r + 8 * hi) * HDIM + nt * 16 + lo] = (half_t)sn;
  }
}

__global__ __launch_bounds__(WGSIZE)
__attribute__((amdgpu_num_vgpr(256)))
void tesnn_kernel(
    const float* __restrict__ input, const float* __restrict__ W1,
    const float* __restrict__ b1, const float* __restrict__ W2,
    const float* __restrict__ b2, const float* __restrict__ W3,
    const float* __restrict__ b3, const float* __restrict__ Wo,
    const float* __restrict__ bo, const float* __restrict__ ce,
    float* __restrict__ out) {
  __shared__ __align__(16) half_t w2h[HDIM * HDIM];
  __shared__ __align__(16) half_t w3h[HDIM * HDIM];
  __shared__ __align__(16) half_t woh[16 * HDIM];
  __shared__ __align__(16) half_t stage[NWAVES][SPW * HDIM];
  __shared__ __align__(16) float  thbuf[NWAVES][3][SPW * HDIM];

  const int tid  = threadIdx.x;
  const int wave = tid >> 5;
  const int lane = tid & 31;
  const int lo   = lane & 15;
  const int hi   = lane >> 4;

  // Weights -> f16 in LDS (spikes are exactly 0/1 in f16; only weights round).
#pragma unroll 4
  for (int i = tid; i < HDIM * HDIM; i += WGSIZE) {
    w2h[i] = (half_t)W2[i];
    w3h[i] = (half_t)W3[i];
  }
  for (int i = tid; i < 16 * HDIM; i += WGSIZE)
    woh[i] = (i < OUTD * HDIM) ? (half_t)Wo[i] : (half_t)0.f;

  half_t* stg = stage[wave];
  float* th1 = &thbuf[wave][0][lane * 32];
  float* th2 = &thbuf[wave][1][lane * 32];
  float* th3 = &thbuf[wave][2][lane * 32];
#pragma unroll
  for (int i = 0; i < 32; ++i) { th1[i] = 0.5f; th2[i] = 0.5f; th3[i] = 0.5f; }
  __syncthreads();

  const int n0 = blockIdx.x * SPWG + wave * SPW;
  float w1v[4], b1v[4], b2v[4], b3v[4];
#pragma unroll
  for (int nt = 0; nt < 4; ++nt) {
    int h = nt * 16 + lo;
    w1v[nt] = W1[h]; b1v[nt] = b1[h]; b2v[nt] = b2[h]; b3v[nt] = b3[h];
  }
  float bov = (lo < OUTD) ? bo[lo] : 0.f;

  float m1[32], m2[32], m3[32];
#pragma unroll
  for (int i = 0; i < 32; ++i) { m1[i] = 0.f; m2[i] = 0.f; m3[i] = 0.f; }
  v8f osum;
#pragma unroll
  for (int i = 0; i < 8; ++i) osum[i] = 0.f;

  const float* xptr = input + (size_t)(n0 + lo) * T_STEPS;
  const float* cptr = ce + (size_t)lo * T_STEPS;

  for (int t = 0; t < T_STEPS; ++t) {
    __builtin_prefetch(xptr + t + 32, 0, 1);
    float xl = xptr[t];
    float cov[4];
    cov[0] = cptr[t];
    cov[1] = cptr[16 * T_STEPS + t];
    cov[2] = cptr[32 * T_STEPS + t];
    cov[3] = cptr[48 * T_STEPS + t];

    // broadcast x across the M dimension of the C layout via ds_bpermute
    float xr[8];
#pragma unroll
    for (int r = 0; r < 8; ++r)
      xr[r] = __int_as_float(
          __builtin_amdgcn_ds_bpermute((r + hi * 8) << 2, __float_as_int(xl)));

    // layer 1 (outer product, elementwise)
#pragma unroll
    for (int nt = 0; nt < 4; ++nt) {
      float hv[8];
#pragma unroll
      for (int r = 0; r < 8; ++r) hv[r] = __builtin_fmaf(xr[r], w1v[nt], b1v[nt]);
      update_tile(m1, th1, hv, cov[nt], stg, lo, hi, nt);
    }
    __builtin_amdgcn_sched_barrier(0);

    // layer 2: h2 = s1 @ W2^T + b2 (WMMA), then state update
    {
      v16h a0 = load_afrag(stg, lo, hi, 0);
      v16h a1 = load_afrag(stg, lo, hi, 1);
#pragma unroll
      for (int nt = 0; nt < 4; ++nt) {
        v8f acc;
#pragma unroll
        for (int i = 0; i < 8; ++i) acc[i] = b2v[nt];
        acc = wmma16(a0, load_bfrag(w2h, lo, hi, nt, 0), acc);
        acc = wmma16(a1, load_bfrag(w2h, lo, hi, nt, 1), acc);
        float hv[8];
#pragma unroll
        for (int r = 0; r < 8; ++r) hv[r] = acc[r];
        update_tile(m2, th2, hv, cov[nt], stg, lo, hi, nt);
      }
    }
    __builtin_amdgcn_sched_barrier(0);

    // layer 3: h3 = s2 @ W3^T + b3
    {
      v16h a0 = load_afrag(stg, lo, hi, 0);
      v16h a1 = load_afrag(stg, lo, hi, 1);
#pragma unroll
      for (int nt = 0; nt < 4; ++nt) {
        v8f acc;
#pragma unroll
        for (int i = 0; i < 8; ++i) acc[i] = b3v[nt];
        acc = wmma16(a0, load_bfrag(w3h, lo, hi, nt, 0), acc);
        acc = wmma16(a1, load_bfrag(w3h, lo, hi, nt, 1), acc);
        float hv[8];
#pragma unroll
        for (int r = 0; r < 8; ++r) hv[r] = acc[r];
        update_tile(m3, th3, hv, cov[nt], stg, lo, hi, nt);
      }
    }
    __builtin_amdgcn_sched_barrier(0);

    // output: osum += s3 @ Wo^T (bias folded in at the end)
    {
      v16h a0 = load_afrag(stg, lo, hi, 0);
      v16h a1 = load_afrag(stg, lo, hi, 1);
      osum = wmma16(a0, load_bfrag(woh, lo, hi, 0, 0), osum);
      osum = wmma16(a1, load_bfrag(woh, lo, hi, 0, 1), osum);
    }
  }

  const float inv_t = 1.f / (float)T_STEPS;
#pragma unroll
  for (int r = 0; r < 8; ++r) {
    if (lo < OUTD) {
      int n = n0 + r + 8 * hi;
      out[(size_t)n * OUTD + lo] = osum[r] * inv_t + bov;
    }
  }
}

extern "C" void kernel_launch(void* const* d_in, const int* in_sizes, int n_in,
                              void* d_out, int out_size, void* d_ws, size_t ws_size,
                              hipStream_t stream) {
  const float* input = (const float*)d_in[0];
  const float* W1 = (const float*)d_in[1];
  const float* b1 = (const float*)d_in[2];
  const float* W2 = (const float*)d_in[3];
  const float* b2 = (const float*)d_in[4];
  const float* W3 = (const float*)d_in[5];
  const float* b3 = (const float*)d_in[6];
  const float* Wo = (const float*)d_in[7];
  const float* bo = (const float*)d_in[8];
  const float* ce = (const float*)d_in[9];
  float* out = (float*)d_out;

  int N = in_sizes[0] / T_STEPS;       // 16384
  dim3 grid(N / SPWG);                 // 256 workgroups of 4 waves
  tesnn_kernel<<<grid, WGSIZE, 0, stream>>>(input, W1, b1, W2, b2, W3, b3, Wo, bo,
                                            ce, out);
}
